// DPoolLSTM_90477781057843
// MI455X (gfx1250) — matrix operand order repaired and checked by reading, not compiled, not run on hardware
//
#include <hip/hip_runtime.h>
#include <hip/hip_bf16.h>

typedef __attribute__((ext_vector_type(16))) _Float16 v16h;
typedef __attribute__((ext_vector_type(8)))  _Float16 v8h;
typedef __attribute__((ext_vector_type(8)))  float    v8f;

#define N_AGENTS 2048
#define KPOOL 288          // 32 (g) + 256 (hp)
#define GPOOL 1024         // 4*256
#define HPOOL 256
#define KMAIN 224          // 64 (emb) + 32 (pooled) + 128 (h)
#define GMAIN 512          // 4*128
#define HMAIN 128

__device__ __forceinline__ float sigf(float x) { return 1.0f / (1.0f + __expf(-x)); }

// ---------------------------------------------------------------------------
// Weight prep: pack [Wih; Whh] (row-major f32, [K,G]) into f16 B-tile swizzle.
// Tile (kt, nt) covers K rows kt*32..+31, cols nt*16..+15. Within a tile,
// lane L holds column (nt*16 + (L&15)), K-halves (L>>4)*16 + kk (kk=0..15),
// stored contiguously: out[(kt*(G/16)+nt)*512 + L*16 + kk].
// Contiguous k order, same as the GEMM A loads (lanes 0-15: K 0..15,
// lanes 16-31: K 16..31, column per lane).
// ---------------------------------------------------------------------------
__global__ void k_prep_B(const float* __restrict__ Wih, const float* __restrict__ Whh,
                         int rows_ih, int K, int G, _Float16* __restrict__ out) {
    int idx = blockIdx.x * blockDim.x + threadIdx.x;
    if (idx >= K * G) return;
    int tile   = idx >> 9;        // /512 halves per tile
    int within = idx & 511;
    int lane   = within >> 4;
    int kk     = within & 15;
    int g16 = G >> 4;
    int kt = tile / g16;
    int nt = tile % g16;
    int k = kt * 32 + ((lane >> 4) << 4) + kk;
    int n = nt * 16 + (lane & 15);
    float w = (k < rows_ih) ? Wih[(size_t)k * G + n]
                            : Whh[(size_t)(k - rows_ih) * G + n];
    out[idx] = (_Float16)w;
}

__global__ void k_zero(unsigned int* p, int nwords) {
    int i = blockIdx.x * blockDim.x + threadIdx.x;
    if (i < nwords) p[i] = 0u;
}

// ---------------------------------------------------------------------------
// Neighbor pooling prep + input embedding (one thread per agent).
// Stages positions & velocities of all 2048 agents in LDS (32 KB), finds the
// 4 nearest neighbors, runs the 4->8 pooling MLP into Xpool cols 0..31, and
// the 2->62 input embedding (vel*4, ReLU) into Xmain cols 0..63 (62,63 = 0).
// ---------------------------------------------------------------------------
__global__ void k_pool_prep(const float* __restrict__ obs1, const float* __restrict__ obs2,
                            const float* __restrict__ W_ie, const float* __restrict__ b_ie,
                            const float* __restrict__ W_pe, const float* __restrict__ b_pe,
                            _Float16* __restrict__ Xpool, _Float16* __restrict__ Xmain) {
    __shared__ float sp[2 * N_AGENTS];
    __shared__ float sv[2 * N_AGENTS];
    int tid = threadIdx.x;
    for (int i = tid; i < N_AGENTS; i += 256) {
        float o1x = obs1[2 * i], o1y = obs1[2 * i + 1];
        float o2x = obs2[2 * i], o2y = obs2[2 * i + 1];
        sp[2 * i] = o2x; sp[2 * i + 1] = o2y;
        sv[2 * i] = o2x - o1x; sv[2 * i + 1] = o2y - o1y;
    }
    __syncthreads();
    int a = blockIdx.x * 256 + tid;
    float px = sp[2 * a], py = sp[2 * a + 1];
    float vx = sv[2 * a], vy = sv[2 * a + 1];

    const float INF = 3.4e38f;
    float d0 = INF, d1 = INF, d2 = INF, d3 = INF;
    int   i0 = 0, i1 = 0, i2 = 0, i3 = 0;
    for (int j = 0; j < N_AGENTS; ++j) {
        if (j == a) continue;
        float dx = sp[2 * j] - px, dy = sp[2 * j + 1] - py;
        float d = dx * dx + dy * dy;                   // monotonic with norm
        if (d < d3) {
            if (d < d0)      { d3=d2;i3=i2; d2=d1;i2=i1; d1=d0;i1=i0; d0=d;i0=j; }
            else if (d < d1) { d3=d2;i3=i2; d2=d1;i2=i1; d1=d;i1=j; }
            else if (d < d2) { d3=d2;i3=i2; d2=d;i2=j; }
            else             { d3=d;i3=j; }
        }
    }
    int nb[4] = {i0, i1, i2, i3};
    #pragma unroll
    for (int nn = 0; nn < 4; ++nn) {
        int j = nb[nn];
        float fpx = sp[2 * j] - px, fpy = sp[2 * j + 1] - py;
        float fvx = sv[2 * j] - vx, fvy = sv[2 * j + 1] - vy;
        #pragma unroll
        for (int o = 0; o < 8; ++o) {
            float g = fpx * W_pe[o] + fpy * W_pe[8 + o] +
                      fvx * W_pe[16 + o] + fvy * W_pe[24 + o] + b_pe[o];
            g = g > 0.f ? g : 0.f;
            Xpool[(size_t)a * KPOOL + nn * 8 + o] = (_Float16)g;
        }
    }
    float sx = 4.f * vx, sy = 4.f * vy;
    for (int o = 0; o < 62; ++o) {
        float e = sx * W_ie[o] + sy * W_ie[62 + o] + b_ie[o];
        e = e > 0.f ? e : 0.f;
        Xmain[(size_t)a * KMAIN + o] = (_Float16)e;
    }
    Xmain[(size_t)a * KMAIN + 62] = (_Float16)0.f;
    Xmain[(size_t)a * KMAIN + 63] = (_Float16)0.f;
}

// ---------------------------------------------------------------------------
// WMMA GEMM: gates[2048, G] = bias + X_f16[2048, KT*32] @ Wsw (B-swizzled).
// Block = 8 waves covering a 128(M) x 64(N) output block; all waves share the
// same 64 columns, so the 4 B tiles per K-step (4 KB) are staged cooperatively
// through LDS (double-buffered, one global_load_b128 + ds_store_b128 per
// thread per step, one barrier per step). The WMMA operand path is then
// short-latency ds_load_b128, while the global load for step kt+1 is issued
// before the WMMAs of step kt and hides its latency behind a full iteration.
// A tiles (per-wave 16x32) stay register double-buffered. LSTM biases
// (bih+bhh, per gate column = per lane) are folded into the accumulator init.
// ---------------------------------------------------------------------------
template <int KT, int G>
__global__ void __launch_bounds__(256) k_gemm_wmma(
        const _Float16* __restrict__ X, const _Float16* __restrict__ Wsw,
        const float* __restrict__ bih, const float* __restrict__ bhh,
        float* __restrict__ gates) {
    constexpr int Krow = KT * 32;
    constexpr int g16  = G / 16;
    constexpr int nblk = G / 64;
    __shared__ __align__(64) _Float16 ldsB[2][4 * 512];   // 2 x 4 KB

    int tid  = threadIdx.x;
    int lane = tid & 31;
    int wv   = tid >> 5;                 // wave in block: 0..7
    int mb   = blockIdx.x / nblk;        // 128-row M block
    int n64  = blockIdx.x % nblk;        // 64-col N block
    int col  = lane & 15;
    int mt   = mb * 8 + wv;              // this wave's 16-row tile

    // A addressing: lane<16 -> row mt*16+lane, K [0..15];
    //               lane>=16 -> row mt*16+lane-16, K [16..31]
    const _Float16* xrow  = X + (size_t)(mt * 16 + col) * Krow + ((lane >> 4) << 4);
    // staging: each thread copies one 16B chunk of the 4KB B panel per K-step
    const _Float16* wbase = Wsw + (size_t)(n64 * 4) * 512 + tid * 8;

    v8f acc[4];
    #pragma unroll
    for (int s = 0; s < 4; ++s) {
        int cc = n64 * 64 + s * 16 + col;
        float b = bih[cc] + bhh[cc];
        #pragma unroll
        for (int r = 0; r < 8; ++r) acc[s][r] = b;
    }

    union Au { v16h v; v8h h[2]; };
    Au a_cur, a_nxt;

    // prologue: stage B(kt=0) into LDS buf0, load A(kt=0) into regs
    *(v8h*)(&ldsB[0][0] + tid * 8) = *(const v8h*)(wbase);
    a_cur.h[0] = *(const v8h*)(xrow);
    a_cur.h[1] = *(const v8h*)(xrow + 8);
    __syncthreads();

    #pragma unroll
    for (int kt = 0; kt < KT; ++kt) {
        const int cur = kt & 1;
        v8h stg;
        if (kt + 1 < KT) {   // issue next-step global loads before this step's WMMAs
            stg        = *(const v8h*)(wbase + (size_t)(kt + 1) * g16 * 512);
            a_nxt.h[0] = *(const v8h*)(xrow + (kt + 1) * 32);
            a_nxt.h[1] = *(const v8h*)(xrow + (kt + 1) * 32 + 8);
        }
        const _Float16* bl = &ldsB[cur][0] + lane * 16;
        #pragma unroll
        for (int s = 0; s < 4; ++s) {
            v16h b = *(const v16h*)(bl + s * 512);      // ds_load path
            acc[s] = __builtin_amdgcn_wmma_f32_16x16x32_f16(
                false, a_cur.v, false, b, (short)0, acc[s], false, false);
        }
        if (kt + 1 < KT) {
            *(v8h*)(&ldsB[cur ^ 1][0] + tid * 8) = stg; // fill other buffer
            a_cur = a_nxt;
            __syncthreads();  // stores visible; buf 'cur' free for kt+2 refill
        }
    }

    // D layout: VGPR r -> rel row (lane>>4)*8 + r, rel col lane&15
    int rbase = mt * 16 + ((lane >> 4) << 3);
    #pragma unroll
    for (int s = 0; s < 4; ++s) {
        int cc = n64 * 64 + s * 16 + col;
        #pragma unroll
        for (int r = 0; r < 8; ++r)
            gates[(size_t)(rbase + r) * G + cc] = acc[s][r];
    }
}

// ---------------------------------------------------------------------------
// LSTM cell elementwise (gate order i,f,g,o). Biases already folded into the
// gates by the GEMM. Writes new c, new h (f32) and packs the new h as f16
// straight into the next step's A-matrix columns.
// ---------------------------------------------------------------------------
__global__ void k_lstm_cell(const float* __restrict__ gates,
                            float* __restrict__ hstate, float* __restrict__ cstate,
                            _Float16* __restrict__ Xdst,
                            int HID, int G, int XK, int xoff) {
    int idx = blockIdx.x * blockDim.x + threadIdx.x;
    int a = idx / HID, j = idx - a * HID;
    if (a >= N_AGENTS) return;
    const float* ga = gates + (size_t)a * G;
    float gi = ga[j];
    float gf = ga[HID + j];
    float gg = ga[2 * HID + j];
    float go = ga[3 * HID + j];
    float c = sigf(gf) * cstate[(size_t)a * HID + j] + sigf(gi) * tanhf(gg);
    cstate[(size_t)a * HID + j] = c;
    float h = sigf(go) * tanhf(c);
    hstate[(size_t)a * HID + j] = h;
    Xdst[(size_t)a * XK + xoff + j] = (_Float16)h;
}

// pooled[2048,32] = hp @ W_h2p + b  -> f16 into Xmain cols 64..95
__global__ void k_pooled(const float* __restrict__ hp, const float* __restrict__ W_h2p,
                         const float* __restrict__ b_h2p, _Float16* __restrict__ Xmain) {
    int idx = blockIdx.x * blockDim.x + threadIdx.x;
    int a = idx >> 5, j = idx & 31;
    if (a >= N_AGENTS) return;
    float s = b_h2p[j];
    const float* hpa = hp + (size_t)a * HPOOL;
    for (int k = 0; k < HPOOL; ++k) s += hpa[k] * W_h2p[k * 32 + j];
    Xmain[(size_t)a * KMAIN + 64 + j] = (_Float16)s;
}

// Hidden2Normal + position integrate; optional feedback into next decoder buf.
__global__ void k_output(const float* __restrict__ h, const float* __restrict__ W_hn,
                         const float* __restrict__ b_hn, const float* __restrict__ base2,
                         float* __restrict__ nrm_out, float* __restrict__ pos_out,
                         float* __restrict__ nxt) {
    int a = blockIdx.x * blockDim.x + threadIdx.x;
    if (a >= N_AGENTS) return;
    float r0 = b_hn[0], r1 = b_hn[1], r2 = b_hn[2], r3 = b_hn[3], r4 = b_hn[4];
    const float* ha = h + (size_t)a * HMAIN;
    for (int j = 0; j < HMAIN; ++j) {
        float hj = ha[j];
        const float* w = W_hn + j * 5;
        r0 += hj * w[0]; r1 += hj * w[1]; r2 += hj * w[2]; r3 += hj * w[3]; r4 += hj * w[4];
    }
    float* n = nrm_out + (size_t)a * 5;
    n[0] = r0; n[1] = r1;
    n[2] = 0.01f + 0.2f * sigf(r2);
    n[3] = 0.01f + 0.2f * sigf(r3);
    n[4] = 0.7f * sigf(r4);
    float px = base2[2 * a] + r0, py = base2[2 * a + 1] + r1;
    pos_out[2 * a] = px; pos_out[2 * a + 1] = py;
    if (nxt) { nxt[2 * a] = px; nxt[2 * a + 1] = py; }
}

// ---------------------------------------------------------------------------
extern "C" void kernel_launch(void* const* d_in, const int* in_sizes, int n_in,
                              void* d_out, int out_size, void* d_ws, size_t ws_size,
                              hipStream_t stream) {
    (void)in_sizes; (void)n_in; (void)out_size; (void)ws_size;
    const float* observed = (const float*)d_in[0];
    const float* W_ie  = (const float*)d_in[4];
    const float* b_ie  = (const float*)d_in[5];
    const float* W_pe  = (const float*)d_in[6];
    const float* b_pe  = (const float*)d_in[7];
    const float* Wih_p = (const float*)d_in[8];
    const float* Whh_p = (const float*)d_in[9];
    const float* bih_p = (const float*)d_in[10];
    const float* bhh_p = (const float*)d_in[11];
    const float* W_h2p = (const float*)d_in[12];
    const float* b_h2p = (const float*)d_in[13];
    const float* Wih_e = (const float*)d_in[14];
    const float* Whh_e = (const float*)d_in[15];
    const float* bih_e = (const float*)d_in[16];
    const float* bhh_e = (const float*)d_in[17];
    const float* Wih_d = (const float*)d_in[18];
    const float* Whh_d = (const float*)d_in[19];
    const float* bih_d = (const float*)d_in[20];
    const float* bhh_d = (const float*)d_in[21];
    const float* W_hn  = (const float*)d_in[22];
    const float* b_hn  = (const float*)d_in[23];

    float* out = (float*)d_out;
    float* normals_out = out;                              // [20][2048][5]
    float* pos_out     = out + (size_t)20 * N_AGENTS * 5;  // [20][2048][2]

    // Workspace carve (all sizes multiples of 256B; state region contiguous)
    char* p = (char*)d_ws;
    float*    gates_p = (float*)p;    p += (size_t)N_AGENTS * GPOOL * 4;  // 8 MB
    float*    gates_m = (float*)p;    p += (size_t)N_AGENTS * GMAIN * 4;  // 4 MB
    _Float16* Wp = (_Float16*)p;      p += (size_t)KPOOL * GPOOL * 2;
    _Float16* We = (_Float16*)p;      p += (size_t)KMAIN * GMAIN * 2;
    _Float16* Wd = (_Float16*)p;      p += (size_t)KMAIN * GMAIN * 2;
    char* zero_base = p;
    _Float16* Xpool = (_Float16*)p;   p += (size_t)N_AGENTS * KPOOL * 2;
    _Float16* Xmain = (_Float16*)p;   p += (size_t)N_AGENTS * KMAIN * 2;
    float* hp = (float*)p;            p += (size_t)N_AGENTS * HPOOL * 4;
    float* cp = (float*)p;            p += (size_t)N_AGENTS * HPOOL * 4;
    float* h  = (float*)p;            p += (size_t)N_AGENTS * HMAIN * 4;
    float* c  = (float*)p;            p += (size_t)N_AGENTS * HMAIN * 4;
    size_t zero_words = (size_t)(p - zero_base) / 4;
    float* posbuf[3];
    for (int i = 0; i < 3; ++i) { posbuf[i] = (float*)p; p += (size_t)N_AGENTS * 2 * 4; }

    // One-time prep: swizzled f16 weights + zero the recurrent state region
    k_prep_B<<<(KPOOL * GPOOL + 255) / 256, 256, 0, stream>>>(Wih_p, Whh_p, 32, KPOOL, GPOOL, Wp);
    k_prep_B<<<(KMAIN * GMAIN + 255) / 256, 256, 0, stream>>>(Wih_e, Whh_e, 96, KMAIN, GMAIN, We);
    k_prep_B<<<(KMAIN * GMAIN + 255) / 256, 256, 0, stream>>>(Wih_d, Whh_d, 96, KMAIN, GMAIN, Wd);
    k_zero<<<(int)((zero_words + 255) / 256), 256, 0, stream>>>((unsigned int*)zero_base, (int)zero_words);

    const int T_ENC = 8;          // T_OBS - 1
    const int NPRED = 12;         // n_predict (compile-time constant for this problem)
    const float* prev2 = nullptr;
    const float* prev1 = nullptr;

    for (int t = 0; t < T_ENC + NPRED; ++t) {
        const float *o1, *o2;
        float* nxt = nullptr;
        if (t < T_ENC) {
            o1 = observed + (size_t)t * N_AGENTS * 2;
            o2 = observed + (size_t)(t + 1) * N_AGENTS * 2;
        } else {
            int k = t - T_ENC;
            if (k == 0) {
                prev2 = observed + (size_t)(T_ENC - 1) * N_AGENTS * 2;
                prev1 = observed + (size_t)T_ENC * N_AGENTS * 2;
            }
            o1 = prev2; o2 = prev1;
            nxt = posbuf[k % 3];
        }

        k_pool_prep<<<N_AGENTS / 256, 256, 0, stream>>>(o1, o2, W_ie, b_ie, W_pe, b_pe, Xpool, Xmain);
        // pool GEMM: (2048/128) M-blocks x (1024/64) N-blocks = 256 blocks
        k_gemm_wmma<KPOOL / 32, GPOOL><<<256, 256, 0, stream>>>(Xpool, Wp, bih_p, bhh_p, gates_p);
        k_lstm_cell<<<(N_AGENTS * HPOOL) / 256, 256, 0, stream>>>(gates_p, hp, cp,
                                                                  Xpool, HPOOL, GPOOL, KPOOL, 32);
        k_pooled<<<(N_AGENTS * 32) / 256, 256, 0, stream>>>(hp, W_h2p, b_h2p, Xmain);
        // main GEMM: (2048/128) x (512/64) = 128 blocks
        k_gemm_wmma<KMAIN / 32, GMAIN><<<128, 256, 0, stream>>>(Xmain, (t < T_ENC) ? We : Wd,
                                             (t < T_ENC) ? bih_e : bih_d,
                                             (t < T_ENC) ? bhh_e : bhh_d, gates_m);
        k_lstm_cell<<<(N_AGENTS * HMAIN) / 256, 256, 0, stream>>>(gates_m,
                                             h, c, Xmain, HMAIN, GMAIN, KMAIN, 96);
        k_output<<<N_AGENTS / 256, 256, 0, stream>>>(h, W_hn, b_hn, o2,
                                             normals_out + (size_t)t * N_AGENTS * 5,
                                             pos_out + (size_t)t * N_AGENTS * 2, nxt);
        if (t >= T_ENC) { prev2 = prev1; prev1 = nxt; }
    }
}
